// UnifiedEngineBlock_61607010894205
// MI455X (gfx1250) — matrix operand order repaired and checked by reading, not compiled, not run on hardware
//
#include <hip/hip_runtime.h>
#include <stdint.h>

// ---------------------------------------------------------------------------
// MI455X / gfx1250 implementation. wave32, WMMA bf16 (f32 accumulate),
// async global->LDS staging (ASYNCcnt) with double-buffered GEMM pipeline.
// ---------------------------------------------------------------------------

typedef __bf16 bf16;
typedef __attribute__((ext_vector_type(16))) __bf16 v16bf;
typedef __attribute__((ext_vector_type(8)))  __bf16 v8bf;
typedef __attribute__((ext_vector_type(8)))  float  v8f;

constexpr int Bb  = 4;
constexpr int Ss  = 2048;
constexpr int Dd  = 1024;
constexpr int Hh  = 4;
constexpr int HDd = Dd / Hh;     // 256
constexpr int Ee  = 4;
constexpr int DFf = Dd * Ee;     // 4096
constexpr int TOK = Bb * Ss;     // 8192

__device__ __forceinline__ v8f wmma_bf16(v16bf a, v16bf b, v8f c) {
  // v_wmma_f32_16x16x32_bf16  (neg_a, A, neg_b, B, c_mod, C, reuse_a, reuse_b)
  return __builtin_amdgcn_wmma_f32_16x16x32_bf16(false, a, false, b, (short)0, c,
                                                 false, false);
}

// Async global->LDS copy, 16B per lane. Tracked with ASYNCcnt, bypasses VGPRs.
__device__ __forceinline__ void async_ld_b128(unsigned lds_off, const void* gaddr) {
  asm volatile("global_load_async_to_lds_b128 %0, %1, off"
               :: "v"(lds_off), "v"(gaddr) : "memory");
}
__device__ __forceinline__ void wait_async0() {
  asm volatile("s_wait_asynccnt 0x0" ::: "memory");
}

__device__ __forceinline__ float gelu_exact(float x) {
  return 0.5f * x * (1.0f + erff(x * 0.70710678118654752f));
}

// Block-wide mean / rstd over D=1024 (256 threads, 8 wave32s).
__device__ __forceinline__ void block_stats(float s, float s2, float* red,
                                            float& mean, float& rstd) {
  #pragma unroll
  for (int o = 16; o > 0; o >>= 1) {
    s  += __shfl_xor(s,  o, 32);
    s2 += __shfl_xor(s2, o, 32);
  }
  int wid = threadIdx.x >> 5, lane = threadIdx.x & 31;
  if (lane == 0) { red[wid] = s; red[8 + wid] = s2; }
  __syncthreads();
  if (threadIdx.x == 0) {
    float a = 0.f, b = 0.f;
    for (int i = 0; i < 8; ++i) { a += red[i]; b += red[8 + i]; }
    red[16] = a; red[17] = b;
  }
  __syncthreads();
  mean = red[16] * (1.0f / Dd);
  float var = red[17] * (1.0f / Dd) - mean * mean;
  rstd = rsqrtf(var + 1e-5f);
}

// ---------------------------------------------------------------------------
// Weight repack: f32 [K,N] row-major  ->  bf16 [N,K] row-major (B^T layout)
// ---------------------------------------------------------------------------
__global__ __launch_bounds__(256) void transpose_cvt_kernel(
    const float* __restrict__ src, bf16* __restrict__ dst, int K, int N) {
  __shared__ float tile[32][33];
  int kBase = blockIdx.y * 32, nBase = blockIdx.x * 32;
  int tx = threadIdx.x, ty = threadIdx.y;  // 32 x 8
  #pragma unroll
  for (int i = 0; i < 4; ++i)
    tile[ty + i * 8][tx] = src[(size_t)(kBase + ty + i * 8) * N + nBase + tx];
  __syncthreads();
  #pragma unroll
  for (int i = 0; i < 4; ++i)
    dst[(size_t)(nBase + ty + i * 8) * K + kBase + tx] = (bf16)tile[tx][ty + i * 8];
}

// ---------------------------------------------------------------------------
// Dual layernorm: xn1 = LN(x; t_ln1), xn2 = LN(x; h_ln)  (bf16 GEMM inputs)
// ---------------------------------------------------------------------------
__global__ __launch_bounds__(256) void ln_dual_kernel(
    const float* __restrict__ x,
    const float* __restrict__ g1, const float* __restrict__ b1,
    const float* __restrict__ g2, const float* __restrict__ b2,
    bf16* __restrict__ xn1, bf16* __restrict__ xn2) {
  __shared__ float red[18];
  size_t t = blockIdx.x;
  int tid = threadIdx.x;
  float4 v = ((const float4*)(x + t * Dd))[tid];
  float s  = v.x + v.y + v.z + v.w;
  float s2 = v.x * v.x + v.y * v.y + v.z * v.z + v.w * v.w;
  float mean, rstd;
  block_stats(s, s2, red, mean, rstd);
  float4 G1 = ((const float4*)g1)[tid], B1 = ((const float4*)b1)[tid];
  float4 G2 = ((const float4*)g2)[tid], B2 = ((const float4*)b2)[tid];
  size_t base = t * Dd + tid * 4;
  float n0 = (v.x - mean) * rstd, n1 = (v.y - mean) * rstd;
  float n2 = (v.z - mean) * rstd, n3 = (v.w - mean) * rstd;
  xn1[base + 0] = (bf16)(n0 * G1.x + B1.x); xn2[base + 0] = (bf16)(n0 * G2.x + B2.x);
  xn1[base + 1] = (bf16)(n1 * G1.y + B1.y); xn2[base + 1] = (bf16)(n1 * G2.y + B2.y);
  xn1[base + 2] = (bf16)(n2 * G1.z + B1.z); xn2[base + 2] = (bf16)(n2 * G2.z + B2.z);
  xn1[base + 3] = (bf16)(n3 * G1.w + B1.w); xn2[base + 3] = (bf16)(n3 * G2.w + B2.w);
}

// ---------------------------------------------------------------------------
// Tiled WMMA GEMM:  C[M,N] = epilogue(A[M,K]_bf16 @ Bt[N,K]_bf16 + bias)
// Block tile 128x128, 8 waves (2x4), each wave 64x32 (8 v8f accumulators).
// Double-buffered LDS filled by async global->LDS engine (ASYNCcnt):
// compute tile i from buf(i&1) while tile i+1 streams into buf(~i&1).
// EPI 0: gelu -> bf16 row-major       EPI 1: res + acc + bias -> f32 row-major
// EPI 2: bf16 head-major [B,H,S,HD]   EPI 3: bf16 transposed  [B,H,HD,S]
// ---------------------------------------------------------------------------
template <int EPI>
__global__ __launch_bounds__(256) void gemm_bf16_wmma(
    const bf16* __restrict__ A, const bf16* __restrict__ Bt,
    const float* __restrict__ bias, const float* __restrict__ res,
    void* __restrict__ out, int Mtot, int Ntot, int Ktot) {
  __shared__ __align__(16) bf16 As[2][128 * 48];  // 32 K + 16 pad per row
  __shared__ __align__(16) bf16 Bs[2][128 * 48];
  const int tid  = threadIdx.x;
  const int lane = tid & 31, wid = tid >> 5;
  const int wm = wid & 1, wn = wid >> 1;            // 2 x 4 wave grid
  const int la = lane & 15, hs = lane >> 4;         // lane-in-16, half-select
  const int mBase = blockIdx.y * 128, nBase = blockIdx.x * 128;

  const unsigned aOff[2] = {(unsigned)(uintptr_t)&As[0][0],
                            (unsigned)(uintptr_t)&As[1][0]};
  const unsigned bOff[2] = {(unsigned)(uintptr_t)&Bs[0][0],
                            (unsigned)(uintptr_t)&Bs[1][0]};

  // Stage 128x32 panels of A and Bt into LDS buffer `buf` via async engine.
  auto stage = [&](int k0, int buf) {
    #pragma unroll
    for (int c = tid; c < 512; c += 256) {
      int r = c >> 2, cc = (c & 3) * 8;
      async_ld_b128(aOff[buf] + (unsigned)(r * 48 + cc) * 2,
                    &A [(size_t)(mBase + r) * Ktot + k0 + cc]);
      async_ld_b128(bOff[buf] + (unsigned)(r * 48 + cc) * 2,
                    &Bt[(size_t)(nBase + r) * Ktot + k0 + cc]);
    }
  };

  v8f acc[4][2] = {};
  const int ntiles = Ktot >> 5;
  stage(0, 0);

  for (int it = 0; it < ntiles; ++it) {
    const int cur = it & 1;
    wait_async0();      // this wave's async fills of buf(cur) have landed
    __syncthreads();    // ... and everyone else's too
    if (it + 1 < ntiles) stage((it + 1) * 32, cur ^ 1);

    // A-frag (16x32 bf16): lanes 0-15 hold K 0-7 / 16-23, lanes 16-31 K 8-15 / 24-31.
    v16bf afrag[4];
    #pragma unroll
    for (int ms = 0; ms < 4; ++ms) {
      int r = wm * 64 + ms * 16 + la;
      union { v16bf v; v8bf h[2]; } u;
      u.h[0] = *(const v8bf*)&As[cur][r * 48 + hs * 8];
      u.h[1] = *(const v8bf*)&As[cur][r * 48 + hs * 8 + 16];
      afrag[ms] = u.v;
    }
    // B-frag (32x16 bf16): lanes 0-15 hold K 0-7 / 16-23, lanes 16-31 K 8-15 / 24-31.
    v16bf bfrag[2];
    #pragma unroll
    for (int ns = 0; ns < 2; ++ns) {
      int r = wn * 32 + ns * 16 + la;
      union { v16bf v; v8bf h[2]; } u;
      u.h[0] = *(const v8bf*)&Bs[cur][r * 48 + hs * 8];
      u.h[1] = *(const v8bf*)&Bs[cur][r * 48 + hs * 8 + 16];
      bfrag[ns] = u.v;
    }
    #pragma unroll
    for (int ms = 0; ms < 4; ++ms)
      #pragma unroll
      for (int ns = 0; ns < 2; ++ns)
        acc[ms][ns] = wmma_bf16(afrag[ms], bfrag[ns], acc[ms][ns]);
  }

  // Epilogue. C-frag: lane holds col la, rows r + 8*hs per VGPR r.
  #pragma unroll
  for (int ms = 0; ms < 4; ++ms)
    #pragma unroll
    for (int ns = 0; ns < 2; ++ns) {
      int col = nBase + wn * 32 + ns * 16 + la;
      float bv = bias[col];
      #pragma unroll
      for (int r = 0; r < 8; ++r) {
        int row = mBase + wm * 64 + ms * 16 + r + hs * 8;
        float v = acc[ms][ns][r] + bv;
        if constexpr (EPI == 0) {
          ((bf16*)out)[(size_t)row * Ntot + col] = (bf16)gelu_exact(v);
        } else if constexpr (EPI == 1) {
          ((float*)out)[(size_t)row * Ntot + col] =
              res[(size_t)row * Ntot + col] + v;
        } else if constexpr (EPI == 2) {
          int b = row / Ss, s = row % Ss, h = col / HDd, hd = col % HDd;
          ((bf16*)out)[((size_t)((b * Hh + h) * Ss + s)) * HDd + hd] = (bf16)v;
        } else {
          int b = row / Ss, s = row % Ss, h = col / HDd, hd = col % HDd;
          ((bf16*)out)[((size_t)((b * Hh + h) * HDd + hd)) * Ss + s] = (bf16)v;
        }
      }
    }
}

// ---------------------------------------------------------------------------
// Hebbian attention: out_i = sum_{j<=i} relu(beta * q_i.k_j) v_j
// q,k: bf16 [B,H,S,HD] ; vt: bf16 [B,H,HD,S] ; out: bf16 [B,S,D] (token-major)
// Block = one (b,h) x 64-query tile. Causal: only kt <= qt key tiles visited.
// ---------------------------------------------------------------------------
__global__ __launch_bounds__(256) void hebbian_attn_kernel(
    const bf16* __restrict__ q, const bf16* __restrict__ k,
    const bf16* __restrict__ vt, const float* __restrict__ betap,
    bf16* __restrict__ attn_out) {
  __shared__ __align__(16) bf16 Qs[64 * 264];   // 64 x 256 (+8 pad)
  __shared__ __align__(16) bf16 affs[64 * 72];  // 64 x 64  (+8 pad)
  const int qt = blockIdx.x, bh = blockIdx.y;
  const int bidx = bh / Hh, hidx = bh % Hh;
  const int tid = threadIdx.x, lane = tid & 31, wid = tid >> 5;
  const int la = lane & 15, hs = lane >> 4;
  const float beta = betap[0];

  // Stage Q tile via async engine (reused for every key tile).
  const bf16* qb = q + ((size_t)bh * Ss + qt * 64) * HDd;
  const unsigned qOff = (unsigned)(uintptr_t)&Qs[0];
  #pragma unroll
  for (int c = tid; c < 64 * 32; c += 256) {
    int r = c >> 5, cc = (c & 31) * 8;
    async_ld_b128(qOff + (unsigned)(r * 264 + cc) * 2, &qb[(size_t)r * HDd + cc]);
  }
  wait_async0();
  __syncthreads();

  const int ma = wid & 3;   // 16-row group for both GEMMs
  const int nh = wid >> 2;  // GEMM1: 32-col half of aff ; GEMM2: 128-feat half
  v8f oacc[8] = {};

  for (int kt = 0; kt <= qt; ++kt) {
    const int kbase = kt * 64;
    // ---- GEMM1: aff(64x64) = Q(64x256) @ K^T, K loaded straight from L2 ----
    v8f c0 = {}, c1 = {};
    const bf16* kb = k + ((size_t)bh * Ss + kbase) * HDd;
    #pragma unroll
    for (int kf = 0; kf < HDd; kf += 32) {
      union { v16bf v; v8bf h[2]; } ua, ub0, ub1;
      int ar = ma * 16 + la;
      ua.h[0] = *(const v8bf*)&Qs[ar * 264 + kf + hs * 8];
      ua.h[1] = *(const v8bf*)&Qs[ar * 264 + kf + hs * 8 + 16];
      const bf16* p0 = kb + (size_t)(nh * 32 + la)      * HDd + kf + hs * 8;
      const bf16* p1 = kb + (size_t)(nh * 32 + 16 + la) * HDd + kf + hs * 8;
      ub0.h[0] = *(const v8bf*)p0; ub0.h[1] = *(const v8bf*)(p0 + 16);
      ub1.h[0] = *(const v8bf*)p1; ub1.h[1] = *(const v8bf*)(p1 + 16);
      c0 = wmma_bf16(ua.v, ub0.v, c0);
      c1 = wmma_bf16(ua.v, ub1.v, c1);
    }
    __syncthreads();  // previous iteration's affs fully consumed
    // beta * relu + causal mask, re-layout C-frag -> A-frag via LDS (bf16).
    #pragma unroll
    for (int ns = 0; ns < 2; ++ns) {
      v8f cv = ns ? c1 : c0;
      int jl = nh * 32 + ns * 16 + la;
      int j = kbase + jl;
      #pragma unroll
      for (int r = 0; r < 8; ++r) {
        int il = ma * 16 + r + hs * 8;
        int i = qt * 64 + il;
        float v = cv[r] * beta;
        v = (j <= i && v > 0.f) ? v : 0.f;
        affs[il * 72 + jl] = (bf16)v;
      }
    }
    __syncthreads();
    // ---- GEMM2: out(64x256) += aff(64x64) @ V(64x256) via Vt rows ----
    const bf16* vb = vt + (size_t)bh * HDd * Ss;
    #pragma unroll
    for (int kk = 0; kk < 64; kk += 32) {
      union { v16bf v; v8bf h[2]; } ua;
      int ar = ma * 16 + la;
      ua.h[0] = *(const v8bf*)&affs[ar * 72 + kk + hs * 8];
      ua.h[1] = *(const v8bf*)&affs[ar * 72 + kk + hs * 8 + 16];
      #pragma unroll
      for (int j = 0; j < 8; ++j) {
        int feat = nh * 128 + j * 16 + la;
        union { v16bf v; v8bf h[2]; } ub;
        const bf16* p = vb + (size_t)feat * Ss + kbase + kk + hs * 8;
        ub.h[0] = *(const v8bf*)p; ub.h[1] = *(const v8bf*)(p + 16);
        oacc[j] = wmma_bf16(ua.v, ub.v, oacc[j]);
      }
    }
  }
  // Write bf16 token-major so the wo GEMM can consume it as A directly.
  #pragma unroll
  for (int j = 0; j < 8; ++j) {
    int feat = nh * 128 + j * 16 + la;
    #pragma unroll
    for (int r = 0; r < 8; ++r) {
      int il = ma * 16 + r + hs * 8;
      int s = qt * 64 + il;
      attn_out[((size_t)(bidx * Ss + s)) * Dd + hidx * HDd + feat] = (bf16)oacc[j][r];
    }
  }
}

// ---------------------------------------------------------------------------
// TRM tail: residual = tpre; xn = LN(tpre); mix = scalar-MLP(xn); out = tpre+mix
// ---------------------------------------------------------------------------
__global__ __launch_bounds__(256) void trm_scalar_mlp_kernel(
    const float* __restrict__ tpre, const float* __restrict__ g,
    const float* __restrict__ bt, const float* __restrict__ sw1,
    const float* __restrict__ sb1, const float* __restrict__ sw2,
    const float* __restrict__ sb2, float* __restrict__ trm_out) {
  __shared__ float red[18];
  size_t t = blockIdx.x;
  int tid = threadIdx.x;
  float4 v = ((const float4*)(tpre + t * Dd))[tid];
  float s  = v.x + v.y + v.z + v.w;
  float s2 = v.x * v.x + v.y * v.y + v.z * v.z + v.w * v.w;
  float mean, rstd;
  block_stats(s, s2, red, mean, rstd);
  float4 G = ((const float4*)g)[tid], Bt = ((const float4*)bt)[tid];
  float w1e[Ee], b1e[Ee], w2e[Ee];
  #pragma unroll
  for (int e = 0; e < Ee; ++e) { w1e[e] = sw1[e]; b1e[e] = sb1[e]; w2e[e] = sw2[e]; }
  float vb2 = sb2[0];
  float xin[4] = {v.x, v.y, v.z, v.w};
  float gg[4] = {G.x, G.y, G.z, G.w}, bb[4] = {Bt.x, Bt.y, Bt.z, Bt.w};
  float4 o;
  float* op = &o.x;
  #pragma unroll
  for (int c = 0; c < 4; ++c) {
    float xn = (xin[c] - mean) * rstd * gg[c] + bb[c];
    float mix = vb2;
    #pragma unroll
    for (int e = 0; e < Ee; ++e) mix += gelu_exact(xn * w1e[e] + b1e[e]) * w2e[e];
    op[c] = xin[c] + mix;
  }
  ((float4*)(trm_out + t * Dd))[tid] = o;
}

// ---------------------------------------------------------------------------
// Final blend + layernorm:  LN(x + sig(a)*trm + (1-sig(a))*heb ; o_ln)
// ---------------------------------------------------------------------------
__global__ __launch_bounds__(256) void blend_ln_kernel(
    const float* __restrict__ x, const float* __restrict__ trm,
    const float* __restrict__ heb, const float* __restrict__ alphap,
    const float* __restrict__ g, const float* __restrict__ bt,
    float* __restrict__ out) {
  __shared__ float red[18];
  size_t t = blockIdx.x;
  int tid = threadIdx.x;
  float bw = 1.0f / (1.0f + expf(-alphap[0]));
  float4 xv = ((const float4*)(x   + t * Dd))[tid];
  float4 tv = ((const float4*)(trm + t * Dd))[tid];
  float4 hv = ((const float4*)(heb + t * Dd))[tid];
  float4 u;
  u.x = xv.x + bw * tv.x + (1.f - bw) * hv.x;
  u.y = xv.y + bw * tv.y + (1.f - bw) * hv.y;
  u.z = xv.z + bw * tv.z + (1.f - bw) * hv.z;
  u.w = xv.w + bw * tv.w + (1.f - bw) * hv.w;
  float s  = u.x + u.y + u.z + u.w;
  float s2 = u.x * u.x + u.y * u.y + u.z * u.z + u.w * u.w;
  float mean, rstd;
  block_stats(s, s2, red, mean, rstd);
  float4 G = ((const float4*)g)[tid], Bt = ((const float4*)bt)[tid];
  float4 o;
  o.x = (u.x - mean) * rstd * G.x + Bt.x;
  o.y = (u.y - mean) * rstd * G.y + Bt.y;
  o.z = (u.z - mean) * rstd * G.z + Bt.z;
  o.w = (u.w - mean) * rstd * G.w + Bt.w;
  ((float4*)(out + t * Dd))[tid] = o;
}

// ---------------------------------------------------------------------------
// Host-side orchestration.
// ---------------------------------------------------------------------------
extern "C" void kernel_launch(void* const* d_in, const int* in_sizes, int n_in,
                              void* d_out, int out_size, void* d_ws, size_t ws_size,
                              hipStream_t stream) {
  const float* x      = (const float*)d_in[0];
  const float* h_ln_g = (const float*)d_in[1];
  const float* h_ln_b = (const float*)d_in[2];
  const float* wq     = (const float*)d_in[3];
  const float* bq     = (const float*)d_in[4];
  const float* wk     = (const float*)d_in[5];
  const float* bk     = (const float*)d_in[6];
  const float* wv     = (const float*)d_in[7];
  const float* bv     = (const float*)d_in[8];
  const float* wo     = (const float*)d_in[9];
  const float* bo     = (const float*)d_in[10];
  const float* beta   = (const float*)d_in[11];
  const float* t_ln1_g= (const float*)d_in[12];
  const float* t_ln1_b= (const float*)d_in[13];
  const float* t_w1   = (const float*)d_in[14];
  const float* t_b1   = (const float*)d_in[15];
  const float* t_w2   = (const float*)d_in[16];
  const float* t_b2   = (const float*)d_in[17];
  const float* t_lns_g= (const float*)d_in[18];
  const float* t_lns_b= (const float*)d_in[19];
  const float* s_w1   = (const float*)d_in[20];
  const float* s_b1   = (const float*)d_in[21];
  const float* s_w2   = (const float*)d_in[22];
  const float* s_b2   = (const float*)d_in[23];
  const float* alpha  = (const float*)d_in[24];
  const float* o_ln_g = (const float*)d_in[25];
  const float* o_ln_b = (const float*)d_in[26];
  float* out = (float*)d_out;

  char* ws = (char*)d_ws;
  const size_t MB = 1ull << 20;
  bf16* wqt  = (bf16*)(ws + 0 * MB);    // 2 MB each (1024x1024 bf16)
  bf16* wkt  = (bf16*)(ws + 2 * MB);
  bf16* wvt  = (bf16*)(ws + 4 * MB);
  bf16* wot  = (bf16*)(ws + 6 * MB);
  bf16* w1t  = (bf16*)(ws + 8 * MB);    // 8 MB  [DF, D]
  bf16* w2t  = (bf16*)(ws + 16 * MB);   // 8 MB  [D, DF]
  bf16* xn1  = (bf16*)(ws + 24 * MB);   // 16 MB [TOK, D]
  bf16* xn2  = (bf16*)(ws + 40 * MB);   // 16 MB [TOK, D]
  // 64 MB region: first h1 [TOK, DF]; after it is consumed, reused as q/k/vt/ao
  bf16* h1   = (bf16*)(ws + 56 * MB);
  bf16* qb   = (bf16*)(ws + 56 * MB);   // 16 MB [B,H,S,HD]
  bf16* kb   = (bf16*)(ws + 72 * MB);   // 16 MB [B,H,S,HD]
  bf16* vtb  = (bf16*)(ws + 88 * MB);   // 16 MB [B,H,HD,S]
  bf16* ao   = (bf16*)(ws + 104 * MB);  // 16 MB [TOK, D]
  float* tpre= (float*)(ws + 120 * MB); // 32 MB f32 (later reused for heb)
  float* hebo= (float*)(ws + 120 * MB);
  float* trmo= (float*)(ws + 152 * MB); // 32 MB f32 ; total = 184 MB

  dim3 tb(32, 8);
  // Weight repacks (f32 [K,N] -> bf16 [N,K])
  transpose_cvt_kernel<<<dim3(Dd / 32,  Dd / 32),  tb, 0, stream>>>(wq,  wqt, Dd,  Dd);
  transpose_cvt_kernel<<<dim3(Dd / 32,  Dd / 32),  tb, 0, stream>>>(wk,  wkt, Dd,  Dd);
  transpose_cvt_kernel<<<dim3(Dd / 32,  Dd / 32),  tb, 0, stream>>>(wv,  wvt, Dd,  Dd);
  transpose_cvt_kernel<<<dim3(Dd / 32,  Dd / 32),  tb, 0, stream>>>(wo,  wot, Dd,  Dd);
  transpose_cvt_kernel<<<dim3(DFf / 32, Dd / 32),  tb, 0, stream>>>(t_w1, w1t, Dd,  DFf);
  transpose_cvt_kernel<<<dim3(Dd / 32,  DFf / 32), tb, 0, stream>>>(t_w2, w2t, DFf, Dd);

  // Dual layernorm -> bf16 GEMM inputs
  ln_dual_kernel<<<TOK, 256, 0, stream>>>(x, t_ln1_g, t_ln1_b, h_ln_g, h_ln_b,
                                          xn1, xn2);

  // ---- TRM path ----
  gemm_bf16_wmma<0><<<dim3(DFf / 128, TOK / 128), 256, 0, stream>>>(
      xn1, w1t, t_b1, nullptr, h1, TOK, DFf, Dd);               // h1 = gelu(.)
  gemm_bf16_wmma<1><<<dim3(Dd / 128, TOK / 128), 256, 0, stream>>>(
      h1, w2t, t_b2, x, tpre, TOK, Dd, DFf);                    // tpre = x + h1@w2
  trm_scalar_mlp_kernel<<<TOK, 256, 0, stream>>>(
      tpre, t_lns_g, t_lns_b, s_w1, s_b1, s_w2, s_b2, trmo);    // trm_out

  // ---- Hebbian path (h1 region now free; reuse for q/k/vt/ao) ----
  gemm_bf16_wmma<2><<<dim3(Dd / 128, TOK / 128), 256, 0, stream>>>(
      xn2, wqt, bq, nullptr, qb, TOK, Dd, Dd);                  // q head-major
  gemm_bf16_wmma<2><<<dim3(Dd / 128, TOK / 128), 256, 0, stream>>>(
      xn2, wkt, bk, nullptr, kb, TOK, Dd, Dd);                  // k head-major
  gemm_bf16_wmma<3><<<dim3(Dd / 128, TOK / 128), 256, 0, stream>>>(
      xn2, wvt, bv, nullptr, vtb, TOK, Dd, Dd);                 // v transposed
  hebbian_attn_kernel<<<dim3(Ss / 64, Bb * Hh), 256, 0, stream>>>(
      qb, kb, vtb, beta, ao);                                   // causal relu attn
  gemm_bf16_wmma<1><<<dim3(Dd / 128, TOK / 128), 256, 0, stream>>>(
      ao, wot, bo, x, hebo, TOK, Dd, Dd);                       // heb = x + ao@wo

  // ---- blend + final layernorm ----
  blend_ln_kernel<<<TOK, 256, 0, stream>>>(x, trmo, hebo, alpha,
                                           o_ln_g, o_ln_b, out);
}
